// CQAttention_51084341018720
// MI455X (gfx1250) — matrix operand order repaired and checked by reading, not compiled, not run on hardware
//
#include <hip/hip_runtime.h>
#include <cstdint>
#include <cstddef>

// Problem constants (from reference): B=24, D=128, N=2048, M=2048
static constexpr int kB = 24;
static constexpr int kD = 128;
static constexpr int kN = 2048;
static constexpr int kM = 2048;
static constexpr float kKeep = 0.9f;          // 1 - DROPOUT_P

typedef __attribute__((ext_vector_type(16))) __bf16 bf16x16;
typedef __attribute__((ext_vector_type(8)))  float  f32x8;
typedef unsigned int u32x4 __attribute__((ext_vector_type(4)));
typedef int          i32x8 __attribute__((ext_vector_type(8)));
typedef int          i32x4 __attribute__((ext_vector_type(4)));

union FragAB { bf16x16 v; unsigned u[8]; };
union PackU  { unsigned u; __bf16 h[2]; };

// ---------------------------------------------------------------------------
// prep kernels: bf16 conversions + bias vectors
// ---------------------------------------------------------------------------
__global__ __launch_bounds__(256) void prep_c_kernel(
    const float* __restrict__ C, const float* __restrict__ Wc,
    const float* __restrict__ Wm, __bf16* __restrict__ Cbf,
    __bf16* __restrict__ Cwt, float* __restrict__ sc)
{
  const int n = blockIdx.x * 256 + threadIdx.x;
  float acc = 0.f;
  #pragma unroll 4
  for (int d = 0; d < kD; ++d) {
    float c = C[(size_t)d * kN + n];
    acc += Wc[d] * c;
    Cbf[(size_t)d * kN + n] = (__bf16)c;
    Cwt[(size_t)n * kD + d] = (__bf16)(c * Wm[d]);
  }
  sc[n] = acc;
}

__global__ __launch_bounds__(256) void prep_q_kernel(
    const float* __restrict__ Q, const float* __restrict__ Wq,
    __bf16* __restrict__ Qbf, float* __restrict__ sq)
{
  const int m = blockIdx.x * 256 + threadIdx.x;
  float acc = 0.f;
  #pragma unroll 4
  for (int d = 0; d < kD; ++d) {
    float q = Q[(size_t)d * kM + m];
    acc += Wq[d] * q;
    Qbf[(size_t)d * kM + m] = (__bf16)q;
  }
  sq[m] = acc;
}

// ---------------------------------------------------------------------------
// TDM helper: issue a 2D tensor_load_to_lds for a 128x32 bf16 tile.
// D# per CDNA5 ISA ch.8: group0 = {count/flags, lds_addr, global_addr, type=2},
// group1 = {data_size=2B, pad(16 DWORDs interval, +1 DWORD) -> LDS row stride
// of 34 bf16 (== LDST), tensor/tile dims, dim0 stride = lda}.
// This toolchain's builtin is the 6-arg form:
//   (u32x4 g0, i32x8 g1, i32x4 g2, i32x4 g3, i32x8 extra, i32 cpol)
// ---------------------------------------------------------------------------
__device__ inline void tdm_load_tile_a(const __bf16* gptr, unsigned ldsOff, int lda)
{
  const unsigned long long ga = (unsigned long long)(uintptr_t)gptr;
  u32x4 g0;
  g0[0] = 1u;                                        // count=1, user mode
  g0[1] = ldsOff;                                    // LDS byte address
  g0[2] = (unsigned)(ga & 0xFFFFFFFFu);              // global_addr[31:0]
  g0[3] = (unsigned)((ga >> 32) & 0x1FFFFFFu)        // global_addr[56:32]
        | (2u << 30);                                // type=2 ("image")
  i32x8 g1;
  g1[0] = (1 << 16)      // data_size = 2 bytes
        | (1 << 20)      // pad_enable
        | (3 << 22)      // pad_interval: 16 DWORDs (one 32-elem bf16 row)
        | (0 << 25);     // pad_amount: 1 DWORD  -> row stride 34 bf16
  g1[1] = (int)(32u << 16);    // tensor_dim0 = 32 (low16 at bits [63:48])
  g1[2] = (int)(128u << 16);   // tensor_dim0 hi=0 | tensor_dim1 = 128
  g1[3] = (int)(32u << 16);    // tensor_dim1 hi=0 | tile_dim0 = 32
  g1[4] = 128;                 // tile_dim1 = 128, tile_dim2 = 0
  g1[5] = lda;                 // tensor_dim0_stride[31:0] (elements)
  g1[6] = 0;                   // stride hi | tensor_dim1_stride lo (unused, 2D)
  g1[7] = 0;
  const i32x4 z4 = {0, 0, 0, 0};
  const i32x8 z8 = {0, 0, 0, 0, 0, 0, 0, 0};
  __builtin_amdgcn_tensor_load_to_lds(g0, g1, z4, z4, z8, 0);
}

// ---------------------------------------------------------------------------
// Generic bf16 WMMA GEMM:  Out[MxN] = A[MxK] (row-major) * B[KxN] (row-major)
// 128x128 tile per workgroup, 8 wave32s, BK=32, v_wmma_f32_16x16x32_bf16.
// A tile staged via the Tensor Data Mover; B tile transposed on fill.
// ---------------------------------------------------------------------------
#define LDST 34   // LDS row stride in bf16 elems (matches TDM pad config)

__global__ __launch_bounds__(256) void gemm_bf16_wmma(
    const __bf16* __restrict__ A, int lda,
    const __bf16* __restrict__ Bm, int ldb,
    int K,
    float* __restrict__ outF, __bf16* __restrict__ outB, int ldc,
    const float* __restrict__ biasRow, const float* __restrict__ biasCol)
{
  __shared__ __bf16 sA[128 * LDST];   // [row][k]   (TDM-filled, padded rows)
  __shared__ __bf16 sB[128 * LDST];   // [col][k]   (transposed on fill)

  const int tid  = threadIdx.x;
  const int lane = tid & 31;
  const int w    = tid >> 5;
  const int wr   = w >> 1;            // wave row 0..3 -> 32 rows each
  const int wc   = w & 1;             // wave col 0..1 -> 64 cols each
  const int rowTile = blockIdx.y * 128;
  const int colTile = blockIdx.x * 128;
  const unsigned sAoff = (unsigned)(size_t)(void*)&sA[0];  // LDS byte offset

  const f32x8 z = {0.f,0.f,0.f,0.f,0.f,0.f,0.f,0.f};
  f32x8 acc[2][4];
  #pragma unroll
  for (int i = 0; i < 2; ++i)
    #pragma unroll
    for (int j = 0; j < 4; ++j) acc[i][j] = z;

  for (int k0 = 0; k0 < K; k0 += 32) {
    // ---- stage A tile via Tensor Data Mover (one wave issues the DMA) ----
    if (w == 0) {
      tdm_load_tile_a(A + (size_t)rowTile * lda + k0, sAoff, lda);
    }
    // ---- stage B tile: 32 K x 128 cols, stored transposed [col][k] ----
    #pragma unroll
    for (int s = 0; s < 8; ++s) {
      int slot = tid + s * 256;
      int k  = slot >> 6;             // 64 dword slots per k-row
      int c2 = slot & 63;             // pair of adjacent columns
      PackU p;
      p.u = *reinterpret_cast<const unsigned*>(
          Bm + (size_t)(k0 + k) * ldb + colTile + c2 * 2);
      sB[(c2 * 2    ) * LDST + k] = p.h[0];
      sB[(c2 * 2 + 1) * LDST + k] = p.h[1];
    }
    // prefetch next K-tile of B into cache (global_prefetch_b8)
    if (k0 + 32 < K) {
      int pk = tid >> 6, pc = tid & 63;
      __builtin_prefetch(Bm + (size_t)(k0 + 32 + pk) * ldb + colTile + pc * 2, 0, 1);
    }
    if (w == 0) __builtin_amdgcn_s_wait_tensorcnt(0);  // A tile landed in LDS
    __syncthreads();

    // ---- fragment loads per ISA 7.12.2 16-bit layouts (wave32) ----
    FragAB afr[2];
    const int half = lane >> 4;
    #pragma unroll
    for (int i = 0; i < 2; ++i) {
      int row = wr * 32 + i * 16 + (lane & 15);
      #pragma unroll
      for (int v = 0; v < 8; ++v) {
        int kk = ((v >> 2) << 4) + (half << 3) + ((v & 3) << 1); // K pair base
        afr[i].u[v] = *reinterpret_cast<const unsigned*>(&sA[row * LDST + kk]);
      }
    }
    FragAB bfr[4];
    const int kbase = half << 4;
    #pragma unroll
    for (int j = 0; j < 4; ++j) {
      int col = wc * 64 + j * 16 + (lane & 15);
      #pragma unroll
      for (int v = 0; v < 8; ++v)
        bfr[j].u[v] = *reinterpret_cast<const unsigned*>(
            &sB[col * LDST + ((v >> 2) << 4) + (half << 3) + ((v & 3) << 1)]);
    }
    #pragma unroll
    for (int i = 0; i < 2; ++i)
      #pragma unroll
      for (int j = 0; j < 4; ++j)
        acc[i][j] = __builtin_amdgcn_wmma_f32_16x16x32_bf16(
            false, afr[i].v, false, bfr[j].v, (short)0, acc[i][j], false, false);
    __syncthreads();
  }

  // ---- epilogue: C/D layout: VGPR e -> row = e + 8*(lane>=16), col = lane%16
  #pragma unroll
  for (int i = 0; i < 2; ++i) {
    #pragma unroll
    for (int j = 0; j < 4; ++j) {
      int rbase = rowTile + wr * 32 + i * 16 + (lane >> 4) * 8;
      int col   = colTile + wc * 64 + j * 16 + (lane & 15);
      #pragma unroll
      for (int e = 0; e < 8; ++e) {
        int row = rbase + e;
        float v = acc[i][j][e];
        if (biasRow) v += biasRow[row] + biasCol[col];
        if (outF) outF[(size_t)row * ldc + col] = v;
        else      outB[(size_t)row * ldc + col] = (__bf16)v;
      }
    }
  }
}

// ---------------------------------------------------------------------------
// Softmax over rows of S (axis m). One workgroup per row.
// ---------------------------------------------------------------------------
__global__ __launch_bounds__(256) void row_softmax_kernel(
    const float* __restrict__ S, __bf16* __restrict__ S1)
{
  __shared__ float red[256];
  const int n = blockIdx.x;
  const int tid = threadIdx.x;
  const float* row = S + (size_t)n * kM;

  float mx = -3.0e38f;
  for (int m = tid; m < kM; m += 256) mx = fmaxf(mx, row[m]);
  red[tid] = mx; __syncthreads();
  for (int s = 128; s > 0; s >>= 1) {
    if (tid < s) red[tid] = fmaxf(red[tid], red[tid + s]);
    __syncthreads();
  }
  const float rmax = red[0]; __syncthreads();

  float sum = 0.f;
  for (int m = tid; m < kM; m += 256) sum += __expf(row[m] - rmax);
  red[tid] = sum; __syncthreads();
  for (int s = 128; s > 0; s >>= 1) {
    if (tid < s) red[tid] += red[tid + s];
    __syncthreads();
  }
  const float inv = 1.0f / red[0];

  for (int m = tid; m < kM; m += 256)
    S1[(size_t)n * kM + m] = (__bf16)(__expf(row[m] - rmax) * inv);
}

// ---------------------------------------------------------------------------
// Softmax over columns of S (axis n); writes transposed S2t[m,n].
// ---------------------------------------------------------------------------
__global__ __launch_bounds__(256) void col_softmax_kernel(
    const float* __restrict__ S, __bf16* __restrict__ S2t)
{
  const int m = blockIdx.x * 256 + threadIdx.x;
  float mx = -3.0e38f;
  for (int n = 0; n < kN; ++n) mx = fmaxf(mx, S[(size_t)n * kM + m]);
  float sum = 0.f;
  for (int n = 0; n < kN; ++n) sum += __expf(S[(size_t)n * kM + m] - mx);
  const float inv = 1.0f / sum;
  for (int n = 0; n < kN; ++n)
    S2t[(size_t)m * kN + n] = (__bf16)(__expf(S[(size_t)n * kM + m] - mx) * inv);
}

// ---------------------------------------------------------------------------
// LDS-tiled bf16 transpose: out[c,r] = in[r,c]
// ---------------------------------------------------------------------------
__global__ __launch_bounds__(256) void transpose_bf16_kernel(
    const __bf16* __restrict__ in, __bf16* __restrict__ outp,
    int rows, int cols)
{
  __shared__ __bf16 t[32][33];
  const int c0 = blockIdx.x * 32, r0 = blockIdx.y * 32;
  for (int i = threadIdx.y; i < 32; i += 8)
    t[i][threadIdx.x] = in[(size_t)(r0 + i) * cols + c0 + threadIdx.x];
  __syncthreads();
  for (int i = threadIdx.y; i < 32; i += 8)
    outp[(size_t)(c0 + i) * rows + r0 + threadIdx.x] = t[threadIdx.x][i];
}

// ---------------------------------------------------------------------------
// Epilogue: out = dropout(concat([C, A, C*A, C*Bt], axis=d)).
// ---------------------------------------------------------------------------
__device__ inline float hash_u01(unsigned long long x) {
  x ^= x >> 33; x *= 0xff51afd7ed558ccdULL;
  x ^= x >> 33; x *= 0xc4ceb9fe1a85ec53ULL;
  x ^= x >> 33;
  return (float)(unsigned)(x >> 40) * (1.0f / 16777216.0f);
}

__global__ __launch_bounds__(256) void epilogue_kernel(
    const float* __restrict__ C, const float* __restrict__ Abuf,
    const float* __restrict__ Btbuf, float* __restrict__ out, int b)
{
  const int idx = blockIdx.x * 256 + threadIdx.x;   // = d*kN + n
  const int d = idx >> 11;
  const int n = idx & (kN - 1);
  const float c  = C[idx];
  const float a  = Abuf[idx];
  const float bt = Btbuf[idx];
  const float vals[4] = { c, a, c * a, c * bt };
  #pragma unroll
  for (int g = 0; g < 4; ++g) {
    size_t o = ((size_t)b * 4 * kD + (size_t)g * kD + d) * kN + n;
    float v = vals[g];
    out[o] = (hash_u01(o) < kKeep) ? v * (1.0f / kKeep) : 0.0f;
  }
}

// ---------------------------------------------------------------------------
// Host launcher: batches run sequentially on `stream` so all per-batch
// intermediates (~52 MB) stay resident in MI455X's 192 MB L2.
// ---------------------------------------------------------------------------
extern "C" void kernel_launch(void* const* d_in, const int* in_sizes, int n_in,
                              void* d_out, int out_size, void* d_ws, size_t ws_size,
                              hipStream_t stream) {
  (void)in_sizes; (void)n_in; (void)out_size; (void)ws_size;
  const float* C = (const float*)d_in[0];
  const float* Q = (const float*)d_in[1];
  const float* W = (const float*)d_in[2];
  float* out = (float*)d_out;

  uint8_t* ws = (uint8_t*)d_ws;
  size_t off = 0;
  auto carve = [&](size_t bytes) -> void* {
    void* p = ws + off;
    off += (bytes + 255) & ~(size_t)255;
    return p;
  };
  float*  Sbuf = (float*) carve((size_t)kN * kM * 4);   // scores, fp32
  __bf16* S1   = (__bf16*)carve((size_t)kN * kM * 2);   // softmax_m
  __bf16* S1t  = (__bf16*)carve((size_t)kN * kM * 2);   // S1 transposed
  __bf16* S2t  = (__bf16*)carve((size_t)kN * kM * 2);   // softmax_n, transposed
  __bf16* Tbuf = (__bf16*)carve((size_t)kN * kN * 2);   // T = S1 * S2^T
  __bf16* Cbf  = (__bf16*)carve((size_t)kD * kN * 2);
  __bf16* Cwt  = (__bf16*)carve((size_t)kN * kD * 2);
  __bf16* Qbf  = (__bf16*)carve((size_t)kD * kM * 2);
  float*  scv  = (float*) carve((size_t)kN * 4);
  float*  sqv  = (float*) carve((size_t)kM * 4);
  float*  Abuf = (float*) carve((size_t)kD * kN * 4);
  float*  Btb  = (float*) carve((size_t)kD * kN * 4);

  for (int b = 0; b < kB; ++b) {
    const float* Cb = C + (size_t)b * kD * kN;
    const float* Qb = Q + (size_t)b * kD * kM;
    const float* Wb = W + (size_t)b * 3 * kD;   // [Wq | Wc | Wm]

    prep_c_kernel<<<kN / 256, 256, 0, stream>>>(Cb, Wb + kD, Wb + 2 * kD,
                                                Cbf, Cwt, scv);
    prep_q_kernel<<<kM / 256, 256, 0, stream>>>(Qb, Wb, Qbf, sqv);

    // S[n,m] = Cwt[n,:] . Qbf[:,m] + sc[n] + sq[m]     (K = 128)
    gemm_bf16_wmma<<<dim3(kM / 128, kN / 128), 256, 0, stream>>>(
        Cwt, kD, Qbf, kM, kD, Sbuf, nullptr, kM, scv, sqv);

    row_softmax_kernel<<<kN, 256, 0, stream>>>(Sbuf, S1);
    col_softmax_kernel<<<kM / 256, 256, 0, stream>>>(Sbuf, S2t);
    transpose_bf16_kernel<<<dim3(kM / 32, kN / 32), dim3(32, 8), 0, stream>>>(
        S1, S1t, kN, kM);

    // T[n,k] = S1[n,:] . S2t[:,k]                      (K = 2048, dominant)
    gemm_bf16_wmma<<<dim3(kN / 128, kN / 128), 256, 0, stream>>>(
        S1, kM, S2t, kN, kM, nullptr, Tbuf, kN, nullptr, nullptr);

    // A[d,n] = Qbf[d,:] . S1t[:,n]                     (K = 2048)
    gemm_bf16_wmma<<<dim3(kN / 128, kD / 128), 256, 0, stream>>>(
        Qbf, kM, S1t, kN, kM, Abuf, nullptr, kN, nullptr, nullptr);

    // Bt[d,k] = Cbf[d,:] . T[:,k]                      (K = 2048)
    gemm_bf16_wmma<<<dim3(kN / 128, kD / 128), 256, 0, stream>>>(
        Cbf, kN, Tbuf, kN, kN, Btb, nullptr, kN, nullptr, nullptr);

    epilogue_kernel<<<(kD * kN) / 256, 256, 0, stream>>>(Cb, Abuf, Btb, out, b);
  }
}
